// GraphEncoderScore_39247411150945
// MI455X (gfx1250) — matrix operand order repaired and checked by reading, not compiled, not run on hardware
//
#include <hip/hip_runtime.h>
#include <hip/hip_bf16.h>
#include <cstddef>

// Problem constants (fixed by the reference)
#define NB   16384      // batch B
#define ND   100        // D
#define NR   100        // R
#define DP   128        // padded D (j axis / i stride)
#define NTL  7          // n-tiles of 16 covering R=100 (7*16=112)
#define NCH  400        // K-chunks of 32: kbig = i*128+j, i in [0,100) -> 100*128/32

typedef __attribute__((ext_vector_type(16))) _Float16 v16h;
typedef __attribute__((ext_vector_type(8)))  _Float16 v8h;
typedef __attribute__((ext_vector_type(8)))  float    v8f;

// ---------------------------------------------------------------------------
// Kernel 1: per-block partial BatchNorm stats of gathered h and t columns.
// 128 blocks x 128 threads (thread = column d). Deterministic (no atomics).
// ---------------------------------------------------------------------------
__global__ __launch_bounds__(128) void stats_partial_kernel(
    const float* __restrict__ feat, const int* __restrict__ hidx,
    const int* __restrict__ tidx, float* __restrict__ part) {
  const int d   = threadIdx.x;          // 0..127 (active d<100)
  const int blk = blockIdx.x;           // 0..127
  const bool act = (d < ND);
  float sh = 0.f, sh2 = 0.f, st = 0.f, st2 = 0.f;
  const int b0 = blk * 128;
  for (int b = b0; b < b0 + 128; ++b) {
    if (act) {
      float x = feat[(size_t)hidx[b] * ND + d];
      float y = feat[(size_t)tidx[b] * ND + d];
      sh += x; sh2 += x * x;
      st += y; st2 += y * y;
    }
  }
  part[(blk * 4 + 0) * DP + d] = sh;
  part[(blk * 4 + 1) * DP + d] = sh2;
  part[(blk * 4 + 2) * DP + d] = st;
  part[(blk * 4 + 3) * DP + d] = st2;
}

// ---------------------------------------------------------------------------
// Kernel 2: fixed-order reduction of the 128 partials -> stats[4][128].
// ---------------------------------------------------------------------------
__global__ __launch_bounds__(512) void stats_reduce_kernel(
    const float* __restrict__ part, float* __restrict__ stats) {
  const int s = threadIdx.x >> 7;       // 0..3
  const int d = threadIdx.x & 127;      // 0..127
  float a = 0.f;
  for (int p = 0; p < 128; ++p) a += part[(p * 4 + s) * DP + d];
  stats[s * DP + d] = a;
}

// ---------------------------------------------------------------------------
// Kernel 3: gather + BatchNorm-normalize + pack to f16, j-padded to 128.
// ---------------------------------------------------------------------------
__global__ __launch_bounds__(128) void normalize_pack_kernel(
    const float* __restrict__ feat, const int* __restrict__ hidx,
    const int* __restrict__ tidx, const float* __restrict__ stats,
    const float* __restrict__ g0, const float* __restrict__ be0,
    const float* __restrict__ g2, const float* __restrict__ be2,
    _Float16* __restrict__ hpad, _Float16* __restrict__ tpad) {
  const int d   = threadIdx.x;
  const int blk = blockIdx.x;
  const float invB = 1.0f / (float)NB;
  float mh = 0.f, rh = 1.f, mt = 0.f, rt = 1.f;
  float gh = 0.f, bh = 0.f, gt = 0.f, bt = 0.f;
  if (d < ND) {
    mh = stats[0 * DP + d] * invB;
    float vh = stats[1 * DP + d] * invB - mh * mh;
    rh = rsqrtf(vh + 1e-5f);
    mt = stats[2 * DP + d] * invB;
    float vt = stats[3 * DP + d] * invB - mt * mt;
    rt = rsqrtf(vt + 1e-5f);
    gh = g0[d]; bh = be0[d]; gt = g2[d]; bt = be2[d];
  }
  const int b0 = blk * 128;
  for (int b = b0; b < b0 + 128; ++b) {
    _Float16 hv = (_Float16)0.0f, tv = (_Float16)0.0f;
    if (d < ND) {
      float x = feat[(size_t)hidx[b] * ND + d];
      float y = feat[(size_t)tidx[b] * ND + d];
      hv = (_Float16)((x - mh) * rh * gh + bh);
      tv = (_Float16)((y - mt) * rt * gt + bt);
    }
    hpad[(size_t)b * DP + d] = hv;
    tpad[(size_t)b * DP + d] = tv;
  }
}

// ---------------------------------------------------------------------------
// Kernel 4: W_mat[r,i,j] = sum_k rel[r,k]*W[k,i,j], written directly in the
// WMMA 16-bit B (32x16) register swizzle:  flat = ((c*7+nt)*32 + L)*16 + e
// where lane L holds column N = L&15 and K = (e>>3)*16 + (L>>4)*8 + (e&7) within the chunk.
// Padding (i>=100 || j>=100 || r>=100) written as 0.
// ---------------------------------------------------------------------------
__global__ __launch_bounds__(256) void build_wmat_kernel(
    const float* __restrict__ rel, const float* __restrict__ W,
    _Float16* __restrict__ bsw) {
  const int id = blockIdx.x * 256 + threadIdx.x;
  if (id >= NCH * NTL * 32 * 16) return;
  const int e  = id & 15;
  const int L  = (id >> 4) & 31;
  const int t  = id >> 9;              // c*7 + nt
  const int nt = t % NTL;
  const int c  = t / NTL;
  const int kk   = ((e >> 3) << 4) | ((L >> 4) << 3) | (e & 7);      // K within chunk (0..31)
  const int kbig = c * 32 + kk;
  const int i = kbig >> 7;
  const int j = kbig & 127;
  const int r = nt * 16 + (L & 15);
  float acc = 0.f;
  if (i < ND && j < ND && r < NR) {
    const float* wp = W + i * ND + j;          // stride D*D over k
    const float* rp = rel + r * ND;
    #pragma unroll 4
    for (int k = 0; k < ND; ++k) acc = fmaf(rp[k], wp[(size_t)k * (ND * ND)], acc);
  }
  bsw[id] = (_Float16)acc;
}

// ---------------------------------------------------------------------------
// Kernel 5: main WMMA GEMM.  score[b,r] = sum_{i,j} (h[b,i]*t[b,j])*Wmat[r,i,j]
// M=16384, N=112(pad of 100), K=12800 (i in [0,100), j padded to 128).
// 128 workgroups x 8 waves; wave owns 16 b-rows x 7 n-tiles (56 f32 accums).
// A-fragment (16x32 f16) built in registers: lane L -> row M=L&15,
// K-halves by L>>4: elements 0..7 = K kb..kb+7, 8..15 = K kb+16..kb+23.
// Sigmoid fused in epilogue.
// ---------------------------------------------------------------------------
__global__ __launch_bounds__(256) void score_gemm_kernel(
    const _Float16* __restrict__ hpad, const _Float16* __restrict__ tpad,
    const _Float16* __restrict__ bsw, float* __restrict__ out) {
  const int lane  = threadIdx.x & 31;
  const int wave  = threadIdx.x >> 5;
  const int b0    = blockIdx.x * 128 + wave * 16;
  const int mrow  = lane & 15;
  const int khalf = lane >> 4;                // 0 or 1
  const _Float16* hrow = hpad + (size_t)(b0 + mrow) * DP;
  const _Float16* trow = tpad + (size_t)(b0 + mrow) * DP;

  v8f acc[NTL];
  #pragma unroll
  for (int nt = 0; nt < NTL; ++nt) {
    v8f z = {0.f, 0.f, 0.f, 0.f, 0.f, 0.f, 0.f, 0.f};
    acc[nt] = z;
  }

  // Hoist the t sub-vectors: for chunk q (j0=q*32), this lane needs
  // t[j0 + khalf*8 .. +7] and t[j0 + khalf*8 + 16 .. +7].  8 x v8h = 32 VGPRs.
  v8h tq[8];
  #pragma unroll
  for (int q = 0; q < 4; ++q) {
    tq[2 * q]     = *(const v8h*)(trow + q * 32 + khalf * 8);
    tq[2 * q + 1] = *(const v8h*)(trow + q * 32 + khalf * 8 + 16);
  }

  const v16h* bb0 = (const v16h*)bsw + lane;

  for (int i = 0; i < ND; ++i) {              // i = h index; 4 chunks each
    const _Float16 hh = hrow[i];
    #pragma unroll
    for (int q = 0; q < 4; ++q) {
      const int c = i * 4 + q;
      v16h a;
      #pragma unroll
      for (int e = 0; e < 8; ++e) {
        a[e]     = tq[2 * q][e]     * hh;     // K = kb + e
        a[e + 8] = tq[2 * q + 1][e] * hh;     // K = kb + 16 + e
      }
      const v16h* bb = bb0 + (size_t)(c * NTL) * 32;
      #pragma unroll
      for (int nt = 0; nt < NTL; ++nt) {
        v16h bf = bb[nt * 32];
        acc[nt] = __builtin_amdgcn_wmma_f32_16x16x32_f16(
            false, a, false, bf, (short)0, acc[nt], false, false);
      }
    }
  }

  // Epilogue: C/D layout -> VGPR v: M = khalf*8 + v, N = lane&15. Fused sigmoid.
  #pragma unroll
  for (int nt = 0; nt < NTL; ++nt) {
    const int col = nt * 16 + mrow;
    if (col < NR) {
      #pragma unroll
      for (int v = 0; v < 8; ++v) {
        const int m = khalf * 8 + v;
        float s = acc[nt][v];
        out[(size_t)(b0 + m) * NR + col] = 1.0f / (1.0f + __expf(-s));
      }
    }
  }
}

// ---------------------------------------------------------------------------
extern "C" void kernel_launch(void* const* d_in, const int* in_sizes, int n_in,
                              void* d_out, int out_size, void* d_ws, size_t ws_size,
                              hipStream_t stream) {
  const float* feature  = (const float*)d_in[0];
  const float* rel_emb  = (const float*)d_in[1];
  const float* W        = (const float*)d_in[2];
  const float* bn0_g    = (const float*)d_in[3];
  const float* bn0_b    = (const float*)d_in[4];
  const float* bn2_g    = (const float*)d_in[5];
  const float* bn2_b    = (const float*)d_in[6];
  const int*   heads    = (const int*)d_in[7];
  const int*   tails    = (const int*)d_in[8];
  float* out = (float*)d_out;

  // Workspace carve-up (~11.4 MB total)
  float*    part  = (float*)d_ws;                       // 128*4*128 f32
  float*    stats = part + 128 * 4 * DP;                // 4*128 f32
  _Float16* hpad  = (_Float16*)(stats + 4 * DP);        // 16384*128 f16
  _Float16* tpad  = hpad + (size_t)NB * DP;             // 16384*128 f16
  _Float16* bsw   = tpad + (size_t)NB * DP;             // 400*7*32*16 f16

  stats_partial_kernel<<<128, 128, 0, stream>>>(feature, heads, tails, part);
  stats_reduce_kernel<<<1, 512, 0, stream>>>(part, stats);
  normalize_pack_kernel<<<128, 128, 0, stream>>>(feature, heads, tails, stats,
                                                 bn0_g, bn0_b, bn2_g, bn2_b,
                                                 hpad, tpad);
  build_wmat_kernel<<<(NCH * NTL * 32 * 16) / 256, 256, 0, stream>>>(rel_emb, W, bsw);
  score_gemm_kernel<<<128, 256, 0, stream>>>(hpad, tpad, bsw, out);
}
